// SimpleAttention_44882408243764
// MI455X (gfx1250) — matrix operand order repaired and checked by reading, not compiled, not run on hardware
//
#include <hip/hip_runtime.h>
#include <stdint.h>

#define BDIM 256

typedef __attribute__((ext_vector_type(16))) __bf16 v16bf;
typedef __attribute__((ext_vector_type(8)))  __bf16 v8bf;
typedef __attribute__((ext_vector_type(8)))  float  v8f;
typedef __attribute__((address_space(3)))    __bf16 lds_bf16;

constexpr int B_ = 4, N_ = 8192, D_ = 1024, H_ = 8, DK_ = 128;

// ---------- bf16 helpers (RNE) ----------
__device__ __forceinline__ uint16_t f2bfu(float f) {
  uint32_t u = __builtin_bit_cast(uint32_t, f);
  u += 0x7FFFu + ((u >> 16) & 1u);
  return (uint16_t)(u >> 16);
}
__device__ __forceinline__ __bf16 f2bf(float f) {
  uint16_t h = f2bfu(f);
  return __builtin_bit_cast(__bf16, h);
}
__device__ __forceinline__ uint32_t pack2(float a, float b) {
  return (uint32_t)f2bfu(a) | ((uint32_t)f2bfu(b) << 16);
}

// ---------- WMMA fragment loaders (CDNA5 16x16x32 bf16 layouts) ----------
// A 16x32: lane l (r=l&15,g=l>>4) holds k = {g*8+0..7, 16+g*8+0..7} of row r.
__device__ __forceinline__ v16bf load_fragA(const __bf16* tile, int ld, int kofs) {
  const int lane = threadIdx.x & 31;
  const int r = lane & 15, g = lane >> 4;
  const __bf16* p = tile + (size_t)r * ld + kofs + g * 8;
  v8bf lo = *(const v8bf*)p;
  v8bf hi = *(const v8bf*)(p + 16);
  return __builtin_shufflevector(lo, hi, 0,1,2,3,4,5,6,7,8,9,10,11,12,13,14,15);
}
// B 32x16 stored as [n][k]: lane l (n=l&15,g=l>>4) holds k = {g*8+0..7, 16+g*8+0..7} of row n.
__device__ __forceinline__ v16bf load_fragB(const __bf16* tile, int ld, int kofs) {
  const int lane = threadIdx.x & 31;
  const int r = lane & 15, g = lane >> 4;
  const __bf16* p = tile + (size_t)r * ld + kofs + g * 8;
  v8bf lo = *(const v8bf*)p;
  v8bf hi = *(const v8bf*)(p + 16);
  return __builtin_shufflevector(lo, hi, 0,1,2,3,4,5,6,7,8,9,10,11,12,13,14,15);
}
__device__ __forceinline__ v8f wmma_bf16(v16bf a, v16bf b, v8f c) {
  return __builtin_amdgcn_wmma_f32_16x16x32_bf16(false, a, false, b, (short)0, c, false, false);
}

// =====================================================================
// Kernel: projection + per-head LayerNorm, output transposed [b,h,d,N] bf16
// grid = B*N/128 blocks, 256 threads (8 waves). Each block: 128 rows, all heads.
// =====================================================================
__global__ __launch_bounds__(256, 1)
void k_proj_ln(const float* __restrict__ X, const float* __restrict__ W,
               const float* __restrict__ bias, const float* __restrict__ gamma,
               const float* __restrict__ beta, __bf16* __restrict__ outT)
{
  constexpr int ALD = 1032;   // 128 x 1024 (+8 pad) bf16  -> 258 KB
  constexpr int WLD = 136;    // 128 x 128  (+8 pad) bf16  ->  34 KB
  __shared__ __bf16 sA[128 * ALD];
  __shared__ __bf16 sW[128 * WLD];
  const int tid = threadIdx.x, lane = tid & 31, wave = tid >> 5;
  const int g = lane >> 4, ln = lane & 15;
  const int b  = blockIdx.x >> 6;            // N/128 = 64 row-blocks per batch
  const int n0 = (blockIdx.x & 63) * 128;

  // Stage 128 input rows (fp32 -> bf16), read exactly once from HBM.
  for (int i4 = tid; i4 < 128 * 256; i4 += BDIM) {
    int row = i4 >> 8, c = (i4 & 255) << 2;
    const float4 v = *(const float4*)(X + ((size_t)(b * N_ + n0 + row)) * D_ + c);
    *(uint2*)(&sA[row * ALD + c]) = make_uint2(pack2(v.x, v.y), pack2(v.z, v.w));
  }
  __syncthreads();

  for (int h = 0; h < H_; ++h) {
    v8f acc[8];
    #pragma unroll
    for (int nt = 0; nt < 8; ++nt) acc[nt] = {};
    for (int kc = 0; kc < 8; ++kc) {           // K chunks of 128
      __syncthreads();
      // stage weight chunk [128 e][128 d] fp32 -> bf16 (L2-resident)
      for (int i2 = tid; i2 < 128 * 64; i2 += BDIM) {
        int e = i2 >> 6, c2 = (i2 & 63) << 1;
        const float2 w = *(const float2*)(W + ((size_t)(h * DK_ + e)) * D_ + kc * 128 + c2);
        *(uint32_t*)(&sW[e * WLD + c2]) = pack2(w.x, w.y);
      }
      __syncthreads();
      #pragma unroll
      for (int ks = 0; ks < 4; ++ks) {
        // batch all fragment loads, then run the 8 WMMAs back-to-back
        v16bf a = load_fragA(sA + wave * 16 * ALD, ALD, kc * 128 + ks * 32);
        v16bf bw[8];
        #pragma unroll
        for (int nt = 0; nt < 8; ++nt)
          bw[nt] = load_fragB(sW + nt * 16 * WLD, WLD, ks * 32);
        #pragma unroll
        for (int nt = 0; nt < 8; ++nt)
          acc[nt] = wmma_bf16(a, bw[nt], acc[nt]);
      }
    }
    // + bias
    #pragma unroll
    for (int nt = 0; nt < 8; ++nt) {
      float bb = bias[h * DK_ + nt * 16 + ln];
      #pragma unroll
      for (int i = 0; i < 8; ++i) acc[nt][i] += bb;
    }
    // LayerNorm over the 128 head columns: rows live in 16-lane halves.
    float mu[8], rs[8];
    #pragma unroll
    for (int i = 0; i < 8; ++i) {
      float s = 0.f, q = 0.f;
      #pragma unroll
      for (int nt = 0; nt < 8; ++nt) { float x = acc[nt][i]; s += x; q += x * x; }
      #pragma unroll
      for (int m = 1; m <= 8; m <<= 1) { s += __shfl_xor(s, m, 32); q += __shfl_xor(q, m, 32); }
      float mean = s * (1.f / 128.f);
      float var  = q * (1.f / 128.f) - mean * mean;
      mu[i] = mean;
      rs[i] = rsqrtf(var + 1e-5f);
    }
    // normalize + affine, write transposed [b,h,d,N] as packed bf16 (16B stores)
    #pragma unroll
    for (int nt = 0; nt < 8; ++nt) {
      int col = nt * 16 + ln;
      float gm = gamma[h * DK_ + col], bt = beta[h * DK_ + col];
      float o[8];
      #pragma unroll
      for (int i = 0; i < 8; ++i) o[i] = (acc[nt][i] - mu[i]) * rs[i] * gm + bt;
      size_t off = ((size_t)((b * H_ + h) * DK_ + col)) * N_ + n0 + wave * 16 + g * 8;
      *(uint4*)(outT + off) = make_uint4(pack2(o[0], o[1]), pack2(o[2], o[3]),
                                         pack2(o[4], o[5]), pack2(o[6], o[7]));
    }
  }
}

// =====================================================================
// Kernel: scores partials  S[b,h] = K^T V over a 1024-row chunk.
// Fragments straight from global (transposed bf16 buffers). No LDS.
// grid = 32 (b,h) * 8 chunks = 256 blocks.
// =====================================================================
__global__ __launch_bounds__(256, 1)
void k_scores(const __bf16* __restrict__ kT, const __bf16* __restrict__ vT,
              float* __restrict__ partial)
{
  const int tid = threadIdx.x, lane = tid & 31, wave = tid >> 5;
  const int g = lane >> 4, ln = lane & 15;
  const int bh = blockIdx.x >> 3, cch = blockIdx.x & 7;
  const int row0 = cch * 1024;
  const __bf16* ka = kT + ((size_t)bh * DK_ + wave * 16) * N_;
  const __bf16* vb = vT + (size_t)bh * DK_ * N_;
  v8f acc[8];
  #pragma unroll
  for (int nt = 0; nt < 8; ++nt) acc[nt] = {};
  for (int r = 0; r < 32; ++r) {
    int kofs = row0 + r * 32;
    v16bf a = load_fragA(ka, N_, kofs);
    v16bf bv[8];
    #pragma unroll
    for (int nt = 0; nt < 8; ++nt)
      bv[nt] = load_fragB(vb + (size_t)(nt * 16) * N_, N_, kofs);
    #pragma unroll
    for (int nt = 0; nt < 8; ++nt)
      acc[nt] = wmma_bf16(a, bv[nt], acc[nt]);
  }
  float* pp = partial + (size_t)cch * 524288 + (size_t)bh * 16384;
  #pragma unroll
  for (int nt = 0; nt < 8; ++nt)
    #pragma unroll
    for (int i = 0; i < 8; ++i)
      pp[(wave * 16 + i + 8 * g) * 128 + nt * 16 + ln] = acc[nt][i];
}

// Fixed-order partial reduction (deterministic, no float atomics).
__global__ void k_reduce(const float* __restrict__ partial, float* __restrict__ scores) {
  int i = blockIdx.x * blockDim.x + threadIdx.x;
  if (i >= 524288) return;
  float s = 0.f;
  #pragma unroll
  for (int c = 0; c < 8; ++c) s += partial[(size_t)c * 524288 + i];
  scores[i] = s;
}

// =====================================================================
// Kernel: W2T[b][e'][dd] = (Wq_seg^T @ (scores/N))^T  (bf16), per (b,h,dd-block)
// grid = 4*8*8 = 256 blocks.
// =====================================================================
__global__ __launch_bounds__(256, 1)
void k_w2t(const float* __restrict__ Wq, const float* __restrict__ scores,
           __bf16* __restrict__ W2T)
{
  constexpr int TLD = 136;
  __shared__ __bf16 sAT[128 * TLD];   // A = Wq^T seg : [dd][d]
  __shared__ __bf16 sBT[128 * TLD];   // B = P^T      : [e][d]
  const int tid = threadIdx.x, lane = tid & 31, wave = tid >> 5;
  const int g = lane >> 4, ln = lane & 15;
  const int b = blockIdx.x >> 6, h = (blockIdx.x >> 3) & 7, mb = blockIdx.x & 7;
  const int bh = b * H_ + h;
  for (int i = tid; i < 128 * 128; i += BDIM) {
    int d = i >> 7, c = i & 127;
    sAT[c * TLD + d] = f2bf(Wq[((size_t)(h * DK_ + d)) * D_ + mb * 128 + c]);
    sBT[c * TLD + d] = f2bf(scores[(size_t)bh * 16384 + d * 128 + c] * (1.f / N_));
  }
  __syncthreads();
  v8f acc[8] = {};
  #pragma unroll
  for (int ks = 0; ks < 4; ++ks) {
    v16bf a = load_fragA(sAT + wave * 16 * TLD, TLD, ks * 32);
    v16bf bp[8];
    #pragma unroll
    for (int nt = 0; nt < 8; ++nt)
      bp[nt] = load_fragB(sBT + nt * 16 * TLD, TLD, ks * 32);
    #pragma unroll
    for (int nt = 0; nt < 8; ++nt)
      acc[nt] = wmma_bf16(a, bp[nt], acc[nt]);
  }
  #pragma unroll
  for (int nt = 0; nt < 8; ++nt) {
    int ep = h * DK_ + nt * 16 + ln;
    size_t off = ((size_t)b * D_ + ep) * D_ + mb * 128 + wave * 16 + g * 8;
    *(uint4*)(W2T + off) = make_uint4(pack2(acc[nt][0], acc[nt][1]), pack2(acc[nt][2], acc[nt][3]),
                                      pack2(acc[nt][4], acc[nt][5]), pack2(acc[nt][6], acc[nt][7]));
  }
}

// bias2[b][e'] = sum_d bq[h,d] * P[b,h,d,e]   (tiny)
__global__ void k_bias2(const float* __restrict__ bq, const float* __restrict__ scores,
                        float* __restrict__ bias2)
{
  const int bh = blockIdx.x, b = bh >> 3, h = bh & 7;
  const int e = threadIdx.x;
  float s = 0.f;
  for (int d = 0; d < 128; ++d)
    s += bq[h * DK_ + d] * scores[(size_t)bh * 16384 + d * 128 + e];
  bias2[(size_t)b * D_ + h * DK_ + e] = s * (1.f / N_);
}

// p_attn output = scores / N
__global__ void k_pattn(const float* __restrict__ scores, float* __restrict__ outp) {
  int i = blockIdx.x * blockDim.x + threadIdx.x;
  if (i < 524288) outp[i] = scores[i] * (1.f / N_);
}

// =====================================================================
// Kernel: att_output = query @ W2[b] + bias2[b].  Query staged once in LDS
// (full K); W2T tiles double-buffered via ASYNC global->LDS DMA (no VGPR
// round-trip), ASYNCcnt-synchronized; one barrier per k-step.
// grid = B*N/128 = 256 blocks.
// =====================================================================
__global__ __launch_bounds__(256, 1)
void k_out(const float* __restrict__ query, const __bf16* __restrict__ W2T,
           const float* __restrict__ bias2, float* __restrict__ out)
{
  constexpr int ALD = 1032, BLD = 40;
  __shared__ __bf16 sA[128 * ALD];       // 258 KB
  __shared__ __bf16 sB[2][128 * BLD];    // 2 x 10 KB
  const int tid = threadIdx.x, lane = tid & 31, wave = tid >> 5;
  const int g = lane >> 4, ln = lane & 15;
  const int b  = blockIdx.x >> 6;
  const int n0 = (blockIdx.x & 63) * 128;

  for (int i4 = tid; i4 < 128 * 256; i4 += BDIM) {
    int row = i4 >> 8, c = (i4 & 255) << 2;
    const float4 v = *(const float4*)(query + ((size_t)(b * N_ + n0 + row)) * D_ + c);
    *(uint2*)(&sA[row * ALD + c]) = make_uint2(pack2(v.x, v.y), pack2(v.z, v.w));
  }
  __syncthreads();

  const int nrow = tid >> 1, hf = tid & 1;
  for (int nb = 0; nb < 8; ++nb) {
    v8f acc[8] = {};
    const __bf16* bsrc = W2T + ((size_t)b * D_ + nb * 128) * D_;
    for (int ks = 0; ks < 32; ++ks) {
      { // async DMA stage of B tile [128 n][32 k] into buffer ks&1 (bf16 copy)
        const __bf16* sp = bsrc + (size_t)nrow * D_ + ks * 32 + hf * 16;
        __bf16* dp = &sB[ks & 1][nrow * BLD + hf * 16];
        uint32_t loff = (uint32_t)(uintptr_t)(lds_bf16*)dp;
        uint64_t ga   = (uint64_t)(uintptr_t)sp;
        asm volatile("global_load_async_to_lds_b128 %0, %1, off"
                     :: "v"(loff), "v"(ga) : "memory");
        asm volatile("global_load_async_to_lds_b128 %0, %1, off offset:16"
                     :: "v"(loff), "v"(ga) : "memory");
      }
      asm volatile("s_wait_asynccnt 0" ::: "memory");
      __syncthreads();
      v16bf a = load_fragA(sA + wave * 16 * ALD, ALD, ks * 32);
      v16bf bw[8];
      #pragma unroll
      for (int nt = 0; nt < 8; ++nt)
        bw[nt] = load_fragB(&sB[ks & 1][nt * 16 * BLD], BLD, 0);
      #pragma unroll
      for (int nt = 0; nt < 8; ++nt)
        acc[nt] = wmma_bf16(a, bw[nt], acc[nt]);
    }
    #pragma unroll
    for (int nt = 0; nt < 8; ++nt) {
      int e = nb * 128 + nt * 16 + ln;
      float bb = bias2[(size_t)b * D_ + e];
      int r0 = n0 + wave * 16 + g * 8;
      float* op = out + ((size_t)(b * N_ + r0)) * D_ + e;
      #pragma unroll
      for (int i = 0; i < 8; ++i) op[(size_t)i * D_] = acc[nt][i] + bb;
    }
  }
}

// =====================================================================
extern "C" void kernel_launch(void* const* d_in, const int* in_sizes, int n_in,
                              void* d_out, int out_size, void* d_ws, size_t ws_size,
                              hipStream_t stream)
{
  (void)in_sizes; (void)n_in; (void)out_size; (void)ws_size;
  const float* query = (const float*)d_in[0];
  const float* key   = (const float*)d_in[1];
  const float* value = (const float*)d_in[2];
  const float* Wq = (const float*)d_in[3];
  const float* bq = (const float*)d_in[4];
  const float* Wk = (const float*)d_in[5];
  const float* bk = (const float*)d_in[6];
  const float* Wv = (const float*)d_in[7];
  const float* bv = (const float*)d_in[8];
  const float* gK = (const float*)d_in[9];
  const float* bK = (const float*)d_in[10];
  const float* gV = (const float*)d_in[11];
  const float* bV = (const float*)d_in[12];
  float* out = (float*)d_out;

  char* ws = (char*)d_ws;
  float*  scores  = (float*)(ws + 0);                       //  2 MB
  float*  partial = (float*)(ws + 2097152ull);              // 16 MB
  __bf16* kT      = (__bf16*)(ws + 18874368ull);            // 64 MB
  __bf16* vT      = (__bf16*)(ws + 85983232ull);            // 64 MB
  __bf16* W2T     = (__bf16*)(ws + 153092096ull);           //  8 MB
  float*  bias2   = (float*)(ws + 161480704ull);            // 16 KB

  k_proj_ln<<<256, 256, 0, stream>>>(key,   Wk, bk, gK, bK, kT);
  k_proj_ln<<<256, 256, 0, stream>>>(value, Wv, bv, gV, bV, vT);
  k_scores <<<256, 256, 0, stream>>>(kT, vT, partial);
  k_reduce <<<2048, 256, 0, stream>>>(partial, scores);
  k_w2t    <<<256, 256, 0, stream>>>(Wq, scores, W2T);
  k_bias2  <<<32, 128, 0, stream>>>(bq, scores, bias2);
  k_pattn  <<<2048, 256, 0, stream>>>(scores, out + (size_t)B_ * N_ * D_);
  k_out    <<<256, 256, 0, stream>>>(query, W2T, bias2, out);
}
